// MultiheadAttention_24781961297983
// MI455X (gfx1250) — matrix operand order repaired and checked by reading, not compiled, not run on hardware
//
#include <hip/hip_runtime.h>

// MHA: B=4, N=2048, E=1024, H=16, D=64 — bf16 WMMA pipeline for gfx1250
// with GLOBAL_LOAD_ASYNC_TO_LDS_B128 double-buffered tile streaming.
#define B_ 4
#define N_ 2048
#define E_ 1024
#define H_ 16
#define D_ 64

typedef __bf16 bf16;
typedef __attribute__((ext_vector_type(4)))  bf16  bf16x4;
typedef __attribute__((ext_vector_type(8)))  bf16  bf16x8;
typedef __attribute__((ext_vector_type(16))) bf16  v16bf;
typedef __attribute__((ext_vector_type(8)))  float v8f;

__device__ __forceinline__ v8f wmma_bf16(v16bf a, v16bf b, v8f c) {
  // D = A(16x32 bf16) * B(32x16 bf16) + C(16x16 f32)
  return __builtin_amdgcn_wmma_f32_16x16x32_bf16(false, a, false, b, (short)0, c,
                                                 false, false);
}

// 16B async copy global -> LDS (ASYNCcnt tracked).  VDST operand carries the
// LDS byte address; generic LDS pointers truncate to the DS offset (ISA 10.2).
__device__ __forceinline__ void async_ld16(const bf16* lds_dst, const bf16* g) {
  unsigned lds_off = (unsigned)(unsigned long long)lds_dst;
  asm volatile("global_load_async_to_lds_b128 %0, %1, off"
               :: "v"(lds_off), "v"((unsigned long long)g)
               : "memory");
}
__device__ __forceinline__ void wait_async0() {
  asm volatile("s_wait_asynccnt 0x0" ::: "memory");
}

// A-matrix fragment (16x32): lane r<16 holds row r, K = {k0..k0+7, k0+16..k0+23};
// lane r+16 holds row r, K = {k0+8..k0+15, k0+24..k0+31}.  Two 16B loads.
__device__ __forceinline__ v16bf frag_a(const bf16* base, int ld, int row, int k0) {
  const int lane = threadIdx.x & 31;
  const int hl = lane >> 4, r = lane & 15;
  const bf16* p = base + (row + r) * ld + k0 + hl * 8;
  bf16x8 lo = *(const bf16x8*)p;
  bf16x8 hi = *(const bf16x8*)(p + 16);
  return __builtin_shufflevector(lo, hi, 0,1,2,3,4,5,6,7,8,9,10,11,12,13,14,15);
}

// B-matrix fragment (32x16) from row-major W[col][k]: lane c<16 holds column c,
// K = {k0..k0+7, k0+16..k0+23}; lane c+16 holds column c, K = {k0+8..k0+15, k0+24..k0+31}.  Two 16B loads.
__device__ __forceinline__ v16bf frag_b(const bf16* base, int ld, int col, int k0) {
  const int lane = threadIdx.x & 31;
  const int hl = lane >> 4, c = lane & 15;
  const bf16* p = base + (col + c) * ld + k0 + hl * 8;
  bf16x8 lo = *(const bf16x8*)p;
  bf16x8 hi = *(const bf16x8*)(p + 16);
  return __builtin_shufflevector(lo, hi, 0,1,2,3,4,5,6,7,8,9,10,11,12,13,14,15);
}

// ---------------------------------------------------------------- fp32 -> bf16
__global__ void cvt_f32_to_bf16_kernel(const float* __restrict__ src,
                                       bf16* __restrict__ dst, int n4) {
  int i = blockIdx.x * blockDim.x + threadIdx.x;
  if (i < n4) {
    const float4 v = ((const float4*)src)[i];
    bf16x4 o;
    o[0] = (bf16)v.x; o[1] = (bf16)v.y; o[2] = (bf16)v.z; o[3] = (bf16)v.w;
    ((bf16x4*)dst)[i] = o;
  }
}

// ------------------------------------------------- 128x128 GEMM tile mainloop
// C[M,N] = A[M,K] * W[N,K]^T.  8 waves: wave = 64(M) x 32(N) = 4x2 WMMA tiles.
// LDS tiles 128x32 bf16 (ld 40 to dodge bank conflicts), double buffered, fed
// by async global->LDS copies: one barrier per K-step, loads overlap WMMA.
__device__ __forceinline__ void stage_tiles_async(
    const bf16* __restrict__ A, const bf16* __restrict__ W,
    int lda, int ldb, int m0, int n0, int k0, bf16* sA, bf16* sB) {
#pragma unroll
  for (int j = 0; j < 2; ++j) {
    int chunk = threadIdx.x + j * 256;            // 512 chunks of 8 bf16
    int row = chunk >> 2;                         // 0..127
    int c8 = (chunk & 3) << 3;                    // 0,8,16,24
    async_ld16(&sA[row * 40 + c8], &A[(size_t)(m0 + row) * lda + k0 + c8]);
    async_ld16(&sB[row * 40 + c8], &W[(size_t)(n0 + row) * ldb + k0 + c8]);
  }
}

__device__ __forceinline__ void gemm_tile_128x128(
    const bf16* __restrict__ A, const bf16* __restrict__ W,
    int lda, int ldb, int K, int m0, int n0,
    bf16 (*sA)[128 * 40], bf16 (*sB)[128 * 40], v8f acc[4][2]) {
  const int wid = threadIdx.x >> 5;
  const int wm = wid >> 2, wn = wid & 3;
  v8f z = {};
#pragma unroll
  for (int mt = 0; mt < 4; ++mt)
#pragma unroll
    for (int nt = 0; nt < 2; ++nt) acc[mt][nt] = z;

  stage_tiles_async(A, W, lda, ldb, m0, n0, 0, sA[0], sB[0]);
  for (int k0 = 0; k0 < K; k0 += 32) {
    const int cur = (k0 >> 5) & 1;
    wait_async0();          // our copies into buf[cur] have landed
    __syncthreads();        // everyone's copies landed; prior reads retired
    if (k0 + 32 < K)
      stage_tiles_async(A, W, lda, ldb, m0, n0, k0 + 32, sA[cur ^ 1], sB[cur ^ 1]);
    v16bf af[4], bfr[2];
#pragma unroll
    for (int mt = 0; mt < 4; ++mt) af[mt] = frag_a(sA[cur], 40, wm * 64 + mt * 16, 0);
#pragma unroll
    for (int nt = 0; nt < 2; ++nt) bfr[nt] = frag_b(sB[cur], 40, wn * 32 + nt * 16, 0);
#pragma unroll
    for (int mt = 0; mt < 4; ++mt)
#pragma unroll
      for (int nt = 0; nt < 2; ++nt)
        acc[mt][nt] = wmma_bf16(af[mt], bfr[nt], acc[mt][nt]);
  }
}

// ------------------------------------------------------------------- QKV GEMM
// qkv = x @ w_qkv^T + b_qkv; scatter Q(*1/sqrt(D)),K -> [B,H,N,D] bf16,
// V -> transposed [B,H,D,N] bf16.
__global__ void __launch_bounds__(256)
qkv_gemm_kernel(const bf16* __restrict__ xbf, const bf16* __restrict__ wbf,
                const float* __restrict__ bias, bf16* __restrict__ qws,
                bf16* __restrict__ kws, bf16* __restrict__ vtws) {
  __shared__ bf16 sA[2][128 * 40];
  __shared__ bf16 sB[2][128 * 40];
  const int m0 = blockIdx.y * 128;
  const int n0 = blockIdx.x * 128;
  v8f acc[4][2];
  gemm_tile_128x128(xbf, wbf, E_, E_, E_, m0, n0, sA, sB, acc);

  const int lane = threadIdx.x & 31;
  const int hl = lane >> 4, r16 = lane & 15;
  const int wid = threadIdx.x >> 5, wm = wid >> 2, wn = wid & 3;
#pragma unroll
  for (int nt = 0; nt < 2; ++nt) {
    const int gcol = n0 + wn * 32 + nt * 16 + r16;     // 0..3071
    const float bv = bias[gcol];
    const int t = gcol >> 10, rem = gcol & 1023;
    const int h = rem >> 6, d = rem & 63;
#pragma unroll
    for (int mt = 0; mt < 4; ++mt)
#pragma unroll
      for (int r = 0; r < 8; ++r) {
        const int gm = m0 + wm * 64 + mt * 16 + r + hl * 8;   // 0..8191
        const float v = acc[mt][nt][r] + bv;
        const int bb = gm >> 11, ns = gm & 2047;
        const size_t hb = (size_t)(bb * H_ + h);
        if (t == 0)      qws[(hb * N_ + ns) * D_ + d] = (bf16)(v * 0.125f);
        else if (t == 1) kws[(hb * N_ + ns) * D_ + d] = (bf16)v;
        else             vtws[(hb * D_ + d) * N_ + ns] = (bf16)v;
      }
  }
}

// ------------------------------------------------------------ flash attention
// One WG per (b,h, 128 q rows); each wave owns a 16-row strip; stream 64-key
// blocks of K / V^T through double-buffered LDS (async copies) with online
// softmax; P converts C-layout -> A-layout through a per-wave LDS strip.
__global__ void __launch_bounds__(256)
attn_kernel(const bf16* __restrict__ qws, const bf16* __restrict__ kws,
            const bf16* __restrict__ vtws, bf16* __restrict__ aows) {
  __shared__ bf16 sK[2][64 * 72];     // keys x D, ld 72
  __shared__ bf16 sV[2][64 * 72];     // D x keys (V^T tile), ld 72
  __shared__ bf16 sP[8][16 * 72];     // per-wave P strip, ld 72
  const int bh = blockIdx.y;          // b*16 + h
  const int q0 = blockIdx.x * 128;
  const int b = bh >> 4, h = bh & 15;
  const int wid = threadIdx.x >> 5;
  const int lane = threadIdx.x & 31;
  const int hl = lane >> 4, r16 = lane & 15;

  const bf16* qbase = qws + ((size_t)bh * N_ + q0 + wid * 16) * D_;
  v16bf aq[2];
  aq[0] = frag_a(qbase, D_, 0, 0);
  aq[1] = frag_a(qbase, D_, 0, 32);

  v8f z = {};
  v8f o[4];
#pragma unroll
  for (int dt = 0; dt < 4; ++dt) o[dt] = z;
  float mrow[8], lrow[8];
#pragma unroll
  for (int r = 0; r < 8; ++r) { mrow[r] = -1e30f; lrow[r] = 0.f; }

  const bf16* kbase = kws + (size_t)bh * N_ * D_;
  const bf16* vbase = vtws + (size_t)bh * D_ * N_;

  // async stage of one 64-key K/V^T block
  auto stage_kv = [&](int n0, bf16* dK, bf16* dV) {
#pragma unroll
    for (int j = 0; j < 2; ++j) {
      int chunk = threadIdx.x + j * 256;          // 512 chunks of 8 bf16
      int row = chunk >> 3;                       // 0..63
      int c8 = (chunk & 7) << 3;                  // 0..56
      async_ld16(&dK[row * 72 + c8], &kbase[(size_t)(n0 + row) * D_ + c8]);
      async_ld16(&dV[row * 72 + c8], &vbase[(size_t)row * N_ + n0 + c8]);
    }
  };

  stage_kv(0, sK[0], sV[0]);
  for (int n0 = 0; n0 < N_; n0 += 64) {
    const int cur = (n0 >> 6) & 1;
    wait_async0();
    __syncthreads();
    if (n0 + 64 < N_) stage_kv(n0 + 64, sK[cur ^ 1], sV[cur ^ 1]);

    // S = (Q * 1/sqrt(D)) @ K^T for 64 keys: 4 tiles of 16 cols, K-dim = D
    v8f s[4];
#pragma unroll
    for (int nt = 0; nt < 4; ++nt) {
      s[nt] = z;
      s[nt] = wmma_bf16(aq[0], frag_b(sK[cur], 72, nt * 16, 0), s[nt]);
      s[nt] = wmma_bf16(aq[1], frag_b(sK[cur], 72, nt * 16, 32), s[nt]);
    }

    // Online softmax: rows live in VGPR index, columns across 16-lane halves
#pragma unroll
    for (int r = 0; r < 8; ++r) {
      float mx = s[0][r];
#pragma unroll
      for (int nt = 1; nt < 4; ++nt) mx = fmaxf(mx, s[nt][r]);
#pragma unroll
      for (int off = 8; off >= 1; off >>= 1) mx = fmaxf(mx, __shfl_xor(mx, off, 32));
      const float mnew = fmaxf(mrow[r], mx);
      const float corr = __expf(mrow[r] - mnew);
      mrow[r] = mnew;
      float rs = 0.f;
#pragma unroll
      for (int nt = 0; nt < 4; ++nt) {
        float pv = __expf(s[nt][r] - mnew);
        s[nt][r] = pv;
        rs += pv;
      }
#pragma unroll
      for (int off = 8; off >= 1; off >>= 1) rs += __shfl_xor(rs, off, 32);
      lrow[r] = lrow[r] * corr + rs;
#pragma unroll
      for (int dt = 0; dt < 4; ++dt) o[dt][r] *= corr;
    }

    // C-layout -> A-layout for P via per-wave LDS strip (in-order DS per wave)
    bf16* pw = sP[wid];
#pragma unroll
    for (int nt = 0; nt < 4; ++nt)
#pragma unroll
      for (int r = 0; r < 8; ++r)
        pw[(r + hl * 8) * 72 + nt * 16 + r16] = (bf16)s[nt][r];
    asm volatile("s_wait_dscnt 0x0" ::: "memory");

    // O += P @ V  (A = P 16xK, B from V^T rows = d columns)
#pragma unroll
    for (int kk = 0; kk < 2; ++kk) {
      v16bf pf = frag_a(pw, 72, 0, kk * 32);
#pragma unroll
      for (int dt = 0; dt < 4; ++dt)
        o[dt] = wmma_bf16(pf, frag_b(sV[cur], 72, dt * 16, kk * 32), o[dt]);
    }
  }

  // Normalize and store to attn_out [B, N, E] bf16
  float inv[8];
#pragma unroll
  for (int r = 0; r < 8; ++r) inv[r] = 1.0f / lrow[r];
  const int nrow0 = q0 + wid * 16 + hl * 8;
#pragma unroll
  for (int dt = 0; dt < 4; ++dt) {
    const int e = h * D_ + dt * 16 + r16;
#pragma unroll
    for (int r = 0; r < 8; ++r) {
      const int n = nrow0 + r;
      aows[((size_t)b * N_ + n) * E_ + e] = (bf16)(o[dt][r] * inv[r]);
    }
  }
}

// ------------------------------------------------------------ out projection
__global__ void __launch_bounds__(256)
proj_gemm_kernel(const bf16* __restrict__ a, const bf16* __restrict__ wbf,
                 const float* __restrict__ bias, float* __restrict__ out) {
  __shared__ bf16 sA[2][128 * 40];
  __shared__ bf16 sB[2][128 * 40];
  const int m0 = blockIdx.y * 128;
  const int n0 = blockIdx.x * 128;
  v8f acc[4][2];
  gemm_tile_128x128(a, wbf, E_, E_, E_, m0, n0, sA, sB, acc);

  const int lane = threadIdx.x & 31;
  const int hl = lane >> 4, r16 = lane & 15;
  const int wid = threadIdx.x >> 5, wm = wid >> 2, wn = wid & 3;
#pragma unroll
  for (int nt = 0; nt < 2; ++nt) {
    const int gcol = n0 + wn * 32 + nt * 16 + r16;
    const float bv = bias[gcol];
#pragma unroll
    for (int mt = 0; mt < 4; ++mt)
#pragma unroll
      for (int r = 0; r < 8; ++r) {
        const int gm = m0 + wm * 64 + mt * 16 + r + hl * 8;
        out[(size_t)gm * E_ + gcol] = acc[mt][nt][r] + bv;
      }
  }
}

// -------------------------------------------------------------------- launch
extern "C" void kernel_launch(void* const* d_in, const int* in_sizes, int n_in,
                              void* d_out, int out_size, void* d_ws, size_t ws_size,
                              hipStream_t stream) {
  (void)in_sizes; (void)n_in; (void)out_size; (void)ws_size;
  const float* x      = (const float*)d_in[0];
  const float* w_qkv  = (const float*)d_in[1];
  const float* b_qkv  = (const float*)d_in[2];
  const float* w_proj = (const float*)d_in[3];
  const float* b_proj = (const float*)d_in[4];
  float* out = (float*)d_out;

  char* p = (char*)d_ws;
  auto take = [&](size_t elems) {
    bf16* q = (bf16*)p;
    p += ((elems * sizeof(bf16)) + 255) & ~(size_t)255;
    return q;
  };
  bf16* xbf   = take((size_t)B_ * N_ * E_);      // 16 MB
  bf16* wqkvb = take((size_t)3 * E_ * E_);       //  6 MB
  bf16* wprob = take((size_t)E_ * E_);           //  2 MB
  bf16* qws   = take((size_t)B_ * H_ * N_ * D_); // 16 MB  [B,H,N,D]
  bf16* kws   = take((size_t)B_ * H_ * N_ * D_); // 16 MB  [B,H,N,D]
  bf16* vtws  = take((size_t)B_ * H_ * D_ * N_); // 16 MB  [B,H,D,N]
  bf16* aows  = take((size_t)B_ * N_ * E_);      // 16 MB  [B,N,E]

  auto cvt = [&](const float* s, bf16* d, size_t n) {
    int n4 = (int)(n / 4);
    cvt_f32_to_bf16_kernel<<<(n4 + 255) / 256, 256, 0, stream>>>(s, d, n4);
  };
  cvt(x, xbf, (size_t)B_ * N_ * E_);
  cvt(w_qkv, wqkvb, (size_t)3 * E_ * E_);
  cvt(w_proj, wprob, (size_t)E_ * E_);

  qkv_gemm_kernel<<<dim3(3 * E_ / 128, B_ * N_ / 128), 256, 0, stream>>>(
      xbf, wqkvb, b_qkv, qws, kws, vtws);
  attn_kernel<<<dim3(N_ / 128, B_ * H_), 256, 0, stream>>>(qws, kws, vtws, aows);
  proj_gemm_kernel<<<dim3(E_ / 128, B_ * N_ / 128), 256, 0, stream>>>(
      aows, wprob, b_proj, out);
}
